// Attention_12730283065661
// MI455X (gfx1250) — matrix operand order repaired and checked by reading, not compiled, not run on hardware
//
#include <hip/hip_runtime.h>
#include <hip/hip_bf16.h>

typedef __attribute__((ext_vector_type(16))) _Float16 v16h;
typedef __attribute__((ext_vector_type(8)))  _Float16 v8h;
typedef __attribute__((ext_vector_type(8)))  float    v8f;

#define HD __device__ __forceinline__

constexpr int B_    = 2;
constexpr int S_    = 2048;
constexpr int NX_   = 1024;
constexpr int H_    = 16;
constexpr int D_    = 64;
constexpr int MTOT  = B_ * S_;     // 4096 rows
constexpr int NQKV  = 3 * NX_;     // 3072

HD v8f zero8() {
  v8f z;
#pragma unroll
  for (int i = 0; i < 8; ++i) z[i] = 0.0f;
  return z;
}

HD v8f wmma16(v16h a, v16h b, v8f c) {
  // D = A(16x32 f16) * B(32x16 f16) + C(16x16 f32)
  return __builtin_amdgcn_wmma_f32_16x16x32_f16(
      /*neg_a=*/false, a, /*neg_b=*/false, b,
      /*c_mod=*/(short)0, c, /*reuse_a=*/false, /*reuse_b=*/false);
}

// A-fragment, 16x32 tile from row-major [*, ld] f16 matrix at (row0, k0) == p.
// Layout: lanes 0-15: row=lane, K = 0..15; lanes 16-31:
// row=lane-16, K = 16..31.
HD v16h load_a_frag(const _Float16* __restrict__ p, int ld, int lane) {
  int r = lane & 15;
  int koff = (lane >> 4) * 16;
  const _Float16* q = p + (size_t)r * ld + koff;
  return *(const v16h*)(q);
}

// B-fragment, 32x16 tile where the source is stored "N-major": mat[N][K]
// (transposed weight; K-matrix of attention [keys, d]; transposed V [d, keys]).
// p points at (n0, k0). Layout used here (same k order as A): lanes 0-15 hold K=0..15 of column
// n0+lane; lanes 16-31 hold K=16..31 of column n0+lane-16.
HD v16h load_b_frag_nmajor(const _Float16* __restrict__ p, int ld, int lane) {
  const _Float16* q = p + (size_t)(lane & 15) * ld + (lane >> 4) * 16;
  return *(const v16h*)(q);   // 32 contiguous bytes per lane
}

// ---------------------------------------------------------------- converters
__global__ __launch_bounds__(256) void k_cvt_f16(const float* __restrict__ in,
                                                 _Float16* __restrict__ out, int n) {
  int i = blockIdx.x * blockDim.x + threadIdx.x;
  if (i < n) out[i] = (_Float16)in[i];
}

// in [K, N] fp32 -> out [N, K] f16
__global__ __launch_bounds__(256) void k_cvt_transpose(const float* __restrict__ in,
                                                       _Float16* __restrict__ out,
                                                       int K, int N) {
  int i = blockIdx.x * blockDim.x + threadIdx.x;
  if (i < K * N) {
    int k = i / N, n = i % N;
    out[(size_t)n * K + k] = (_Float16)in[i];
  }
}

// ------------------------------------------------------------------ QKV GEMM
// qkv = x16[4096,1024] @ w_attn. q,k scattered to [B*H, S, D]; v scattered
// TRANSPOSED to [B*H, D, S] so the attention P@V B-fragments are contiguous.
__global__ __launch_bounds__(128) void k_qkv_gemm(
    const _Float16* __restrict__ x16,      // [MTOT, NX]
    const _Float16* __restrict__ wt,       // [NQKV, NX] (transposed weight)
    const float* __restrict__ bias,        // [NQKV]
    _Float16* __restrict__ qb, _Float16* __restrict__ kb,
    _Float16* __restrict__ vb) {
  int lane = threadIdx.x & 31;
  int wave = threadIdx.x >> 5;
  int m0 = blockIdx.x * 16;
  int n0 = (blockIdx.y * 4 + wave) * 64;

  const _Float16* abase = x16 + (size_t)m0 * NX_;
  const _Float16* bbase = wt + (size_t)n0 * NX_;

  v8f acc[4];
#pragma unroll
  for (int t = 0; t < 4; ++t) acc[t] = zero8();

  // software-pipelined, 2x-unrolled K loop: two buffer sets alternate in
  // place (no register-copy rotation); loads for the next 32-step issue
  // before the WMMAs that consume the previous one.
  v16h a0, a1, b0[4], b1[4];
  a0 = load_a_frag(abase, NX_, lane);
#pragma unroll
  for (int t = 0; t < 4; ++t)
    b0[t] = load_b_frag_nmajor(bbase + (size_t)t * 16 * NX_, NX_, lane);

  for (int kk = 0; kk < NX_; kk += 64) {
    __builtin_prefetch(abase + kk + 128, 0, 1);
    __builtin_prefetch(bbase + kk + 128, 0, 1);
    a1 = load_a_frag(abase + kk + 32, NX_, lane);
#pragma unroll
    for (int t = 0; t < 4; ++t)
      b1[t] = load_b_frag_nmajor(bbase + (size_t)t * 16 * NX_ + kk + 32, NX_, lane);
#pragma unroll
    for (int t = 0; t < 4; ++t) acc[t] = wmma16(a0, b0[t], acc[t]);
    if (kk + 64 < NX_) {
      a0 = load_a_frag(abase + kk + 64, NX_, lane);
#pragma unroll
      for (int t = 0; t < 4; ++t)
        b0[t] = load_b_frag_nmajor(bbase + (size_t)t * 16 * NX_ + kk + 64, NX_, lane);
    }
#pragma unroll
    for (int t = 0; t < 4; ++t) acc[t] = wmma16(a1, b1[t], acc[t]);
  }

  int hi = lane >> 4, nl = lane & 15;
#pragma unroll
  for (int t = 0; t < 4; ++t) {
    int c = n0 + t * 16 + nl;             // global output column in [0, 3072)
    float bv = bias[c];
    int sel = c >> 10;                    // 0=q 1=k 2=v
    int hh  = (c & (NX_ - 1)) >> 6;       // head
    int dd  = c & (D_ - 1);               // dim within head
#pragma unroll
    for (int r = 0; r < 8; ++r) {
      int row = m0 + r + 8 * hi;          // global token row in [0, 4096)
      int bb = row >> 11;                 // batch (S=2048)
      int ss = row & (S_ - 1);
      _Float16 val = (_Float16)(acc[t][r] + bv);
      size_t bh = (size_t)bb * H_ + hh;
      if (sel == 0)      qb[(bh * S_ + ss) * D_ + dd] = val;
      else if (sel == 1) kb[(bh * S_ + ss) * D_ + dd] = val;
      else               vb[(bh * D_ + dd) * S_ + ss] = val;   // transposed
    }
  }
}

// ------------------------------------------------------------ flash attention
// One wave handles one 16-row q tile for one (b, h): full D=64 accumulator.
__global__ __launch_bounds__(128) void k_attn(
    const _Float16* __restrict__ qb,       // [B*H, S, D]
    const _Float16* __restrict__ kb,       // [B*H, S, D]
    const _Float16* __restrict__ vb,       // [B*H, D, S] (transposed)
    _Float16* __restrict__ a16 /* [MTOT, NX] */) {
  __shared__ __align__(32) _Float16 pbuf[4][16 * 32];
  int lane = threadIdx.x & 31;
  int wave = threadIdx.x >> 5;
  int tile = blockIdx.x * 4 + wave;       // 0 .. B*H*(S/16)-1
  int qt = tile & (S_ / 16 - 1);
  int bh = tile / (S_ / 16);
  int qbase = qt * 16;
  int bb = bh / H_, hh = bh % H_;

  const _Float16* Q = qb + (size_t)bh * S_ * D_;
  const _Float16* K = kb + (size_t)bh * S_ * D_;
  const _Float16* V = vb + (size_t)bh * D_ * S_;

  // Q fragments for K-dim steps 0 and 32 (D = 64), reused across all key blocks
  v16h aq0 = load_a_frag(Q + (size_t)qbase * D_ + 0, D_, lane);
  v16h aq1 = load_a_frag(Q + (size_t)qbase * D_ + 32, D_, lane);

  v8f o[4];
#pragma unroll
  for (int t = 0; t < 4; ++t) o[t] = zero8();
  float m[8], l[8];
#pragma unroll
  for (int r = 0; r < 8; ++r) { m[r] = -1e30f; l[r] = 0.0f; }

  int hi = lane >> 4, nl = lane & 15;
  const float scale = 0.125f;             // 1/sqrt(64)
  _Float16* pl = pbuf[wave];

  int nblocks = (qbase >> 5) + 1;         // 32-key blocks covering keys <= qbase+15

  auto load_bk = [&](v16h* bk, int k0) {
#pragma unroll
    for (int u = 0; u < 2; ++u) {         // key sub-tiles (0..15 / 16..31)
      bk[2 * u]     = load_b_frag_nmajor(K + (size_t)(k0 + u * 16) * D_ + 0, D_, lane);
      bk[2 * u + 1] = load_b_frag_nmajor(K + (size_t)(k0 + u * 16) * D_ + 32, D_, lane);
    }
  };

  // process one 32-key block: scores -> mask -> online softmax -> P@V.
  // V fragments are loaded right after the score WMMAs; they are consumed
  // only after the softmax VALU work, so the latency overlaps naturally.
  auto process = [&](int k0, const v16h* bk) {
    v8f s0 = zero8(), s1 = zero8();
    s0 = wmma16(aq0, bk[0], s0);
    s0 = wmma16(aq1, bk[1], s0);
    s1 = wmma16(aq0, bk[2], s1);
    s1 = wmma16(aq1, bk[3], s1);

    v16h bv[4];
#pragma unroll
    for (int t = 0; t < 4; ++t)
      bv[t] = load_b_frag_nmajor(V + (size_t)(t * 16) * S_ + k0, S_, lane);

    // scale + causal mask
#pragma unroll
    for (int r = 0; r < 8; ++r) {
      int qrow = qbase + r + 8 * hi;
      float v0 = s0[r] * scale, v1 = s1[r] * scale;
      if (k0 + nl > qrow)      v0 = -1e30f;
      if (k0 + 16 + nl > qrow) v1 = -1e30f;
      s0[r] = v0; s1[r] = v1;
    }
    // online softmax (row reductions across the 16-lane half)
#pragma unroll
    for (int r = 0; r < 8; ++r) {
      float mx = fmaxf(s0[r], s1[r]);
#pragma unroll
      for (int off = 1; off < 16; off <<= 1) mx = fmaxf(mx, __shfl_xor(mx, off, 32));
      float nm = fmaxf(m[r], mx);
      float corr = __expf(m[r] - nm);
      m[r] = nm;
      float p0 = __expf(s0[r] - nm), p1 = __expf(s1[r] - nm);
      s0[r] = p0; s1[r] = p1;
      float sum = p0 + p1;
#pragma unroll
      for (int off = 1; off < 16; off <<= 1) sum += __shfl_xor(sum, off, 32);
      l[r] = l[r] * corr + sum;
#pragma unroll
      for (int t = 0; t < 4; ++t) o[t][r] *= corr;
    }
    // reshape P (C layout) -> A layout via per-wave LDS tile [16][32]
#pragma unroll
    for (int r = 0; r < 8; ++r) {
      pl[(r + 8 * hi) * 32 + nl]      = (_Float16)s0[r];
      pl[(r + 8 * hi) * 32 + 16 + nl] = (_Float16)s1[r];
    }
    v16h ap = load_a_frag(pl, 32, lane);  // same-wave DS ops are in-order
    // O += P @ V for the 4 d-tiles
#pragma unroll
    for (int t = 0; t < 4; ++t) o[t] = wmma16(ap, bv[t], o[t]);
  };

  // pair-unrolled key-block loop: K-fragment double-buffer alternates in
  // place between bk0 and bk1 (no register-copy rotation).
  v16h bk0[4], bk1[4];
  load_bk(bk0, 0);
  int jb = 0;
  for (; jb + 1 < nblocks; jb += 2) {
    load_bk(bk1, (jb + 1) * 32);
    process(jb * 32, bk0);
    if (jb + 2 < nblocks) load_bk(bk0, (jb + 2) * 32);
    process((jb + 1) * 32, bk1);
  }
  if (jb < nblocks) process(jb * 32, bk0);

  // ---- finalize: O / l, merge heads into a16 [B, S, NX] at column h*64+d
#pragma unroll
  for (int r = 0; r < 8; ++r) {
    float inv = 1.0f / l[r];
    int ss = qbase + r + 8 * hi;
    size_t orow = ((size_t)bb * S_ + ss) * NX_ + (size_t)hh * D_;
#pragma unroll
    for (int t = 0; t < 4; ++t)
      a16[orow + t * 16 + nl] = (_Float16)(o[t][r] * inv);
  }
}

// ----------------------------------------------------------------- Proj GEMM
__global__ __launch_bounds__(128) void k_proj_gemm(
    const _Float16* __restrict__ a16,     // [MTOT, NX]
    const _Float16* __restrict__ wt,      // [NX, NX] (transposed weight)
    const float* __restrict__ bias,       // [NX]
    float* __restrict__ out) {            // [MTOT, NX] fp32
  int lane = threadIdx.x & 31;
  int wave = threadIdx.x >> 5;
  int m0 = blockIdx.x * 16;
  int n0 = (blockIdx.y * 4 + wave) * 64;

  const _Float16* abase = a16 + (size_t)m0 * NX_;
  const _Float16* bbase = wt + (size_t)n0 * NX_;

  v8f acc[4];
#pragma unroll
  for (int t = 0; t < 4; ++t) acc[t] = zero8();

  v16h a0, a1, b0[4], b1[4];
  a0 = load_a_frag(abase, NX_, lane);
#pragma unroll
  for (int t = 0; t < 4; ++t)
    b0[t] = load_b_frag_nmajor(bbase + (size_t)t * 16 * NX_, NX_, lane);

  for (int kk = 0; kk < NX_; kk += 64) {
    __builtin_prefetch(abase + kk + 128, 0, 1);
    __builtin_prefetch(bbase + kk + 128, 0, 1);
    a1 = load_a_frag(abase + kk + 32, NX_, lane);
#pragma unroll
    for (int t = 0; t < 4; ++t)
      b1[t] = load_b_frag_nmajor(bbase + (size_t)t * 16 * NX_ + kk + 32, NX_, lane);
#pragma unroll
    for (int t = 0; t < 4; ++t) acc[t] = wmma16(a0, b0[t], acc[t]);
    if (kk + 64 < NX_) {
      a0 = load_a_frag(abase + kk + 64, NX_, lane);
#pragma unroll
      for (int t = 0; t < 4; ++t)
        b0[t] = load_b_frag_nmajor(bbase + (size_t)t * 16 * NX_ + kk + 64, NX_, lane);
    }
#pragma unroll
    for (int t = 0; t < 4; ++t) acc[t] = wmma16(a1, b1[t], acc[t]);
  }

  int hi = lane >> 4, nl = lane & 15;
#pragma unroll
  for (int t = 0; t < 4; ++t) {
    int c = n0 + t * 16 + nl;
    float bv = bias[c];
#pragma unroll
    for (int r = 0; r < 8; ++r) {
      int row = m0 + r + 8 * hi;
      out[(size_t)row * NX_ + c] = acc[t][r] + bv;
    }
  }
}

// -------------------------------------------------------------------- launch
extern "C" void kernel_launch(void* const* d_in, const int* in_sizes, int n_in,
                              void* d_out, int out_size, void* d_ws, size_t ws_size,
                              hipStream_t stream) {
  const float* x      = (const float*)d_in[0];   // [B, S, NX]
  const float* w_attn = (const float*)d_in[1];   // [NX, 3NX]
  const float* b_attn = (const float*)d_in[2];   // [3NX]
  const float* w_proj = (const float*)d_in[3];   // [NX, NX]
  const float* b_proj = (const float*)d_in[4];   // [NX]
  float* out = (float*)d_out;                    // [B, S, NX] fp32

  char* ws = (char*)d_ws;
  size_t off = 0;
  auto carve = [&](size_t bytes) -> void* {
    void* p = ws + off;
    off = (off + bytes + 255) & ~(size_t)255;
    return p;
  };
  _Float16* x16   = (_Float16*)carve((size_t)MTOT * NX_ * 2);        // 8 MB
  _Float16* wta   = (_Float16*)carve((size_t)NQKV * NX_ * 2);        // 6 MB
  _Float16* wtp   = (_Float16*)carve((size_t)NX_ * NX_ * 2);         // 2 MB
  _Float16* qbuf  = (_Float16*)carve((size_t)B_ * H_ * S_ * D_ * 2); // 8 MB
  _Float16* kbuf  = (_Float16*)carve((size_t)B_ * H_ * S_ * D_ * 2); // 8 MB
  _Float16* vbuf  = (_Float16*)carve((size_t)B_ * H_ * S_ * D_ * 2); // 8 MB
  _Float16* abuf  = (_Float16*)carve((size_t)MTOT * NX_ * 2);        // 8 MB

  // 1) fp32 -> f16 conversions / weight transposes
  {
    int n = MTOT * NX_;
    k_cvt_f16<<<(n + 255) / 256, 256, 0, stream>>>(x, x16, n);
  }
  {
    int n = NX_ * NQKV;
    k_cvt_transpose<<<(n + 255) / 256, 256, 0, stream>>>(w_attn, wta, NX_, NQKV);
  }
  {
    int n = NX_ * NX_;
    k_cvt_transpose<<<(n + 255) / 256, 256, 0, stream>>>(w_proj, wtp, NX_, NX_);
  }

  // 2) QKV projection: [4096,1024] x [1024,3072] -> q/k [BH,S,D], v [BH,D,S]
  k_qkv_gemm<<<dim3(MTOT / 16, NQKV / 64 / 4), 128, 0, stream>>>(
      x16, wta, b_attn, qbuf, kbuf, vbuf);

  // 3) flash attention: one wave per 16-row q tile, 4 waves per block
  {
    int tiles = B_ * H_ * (S_ / 16);   // 4096
    k_attn<<<tiles / 4, 128, 0, stream>>>(qbuf, kbuf, vbuf, abuf);
  }

  // 4) output projection -> fp32 d_out
  k_proj_gemm<<<dim3(MTOT / 16, NX_ / 64 / 4), 128, 0, stream>>>(
      abuf, wtp, b_proj, out);
}
